// MultiHeadCrossAttention_91070486544730
// MI455X (gfx1250) — matrix operand order repaired and checked by reading, not compiled, not run on hardware
//
#include <hip/hip_runtime.h>

// ---------------------------------------------------------------------------
// MultiHeadCrossAttention forward for MI455X (gfx1250), wave32 + WMMA bf16.
//   1) cvt:  query/key_value f32 -> bf16
//   2) gemm: Q/K/V projections (bf16 out), WMMA 16x16x32 bf16, f32 accum
//   3) flash attention per (b, h, 64 q-rows), online softmax
//   4) gemm: out = O@Wo + bo (f32 out)
// K/A tiles staged with GLOBAL_LOAD_ASYNC_TO_LDS when available (ASYNCcnt).
// ---------------------------------------------------------------------------

typedef __attribute__((ext_vector_type(16))) __bf16 v16bf;
typedef __attribute__((ext_vector_type(8)))  float  v8f;
typedef __attribute__((ext_vector_type(4)))  int    v4i;

#define DEV static __device__ __forceinline__

constexpr int HID   = 256;
constexpr int NH    = 4;
constexpr int DH    = 64;
constexpr int SEQ   = 4096;
constexpr int BATCH = 2;
constexpr int MROWS = BATCH * SEQ;  // 8192

// --- async global->LDS copy (16B per lane), guarded by builtin presence ----
#if defined(__AMDGCN__) && __has_builtin(__builtin_amdgcn_global_load_async_to_lds_b128)
#define USE_ASYNC_LDS 1
typedef __attribute__((address_space(1))) v4i* gptr_v4i;  // global (AS1)
typedef __attribute__((address_space(3))) v4i* lptr_v4i;  // LDS (AS3)
#else
#define USE_ASYNC_LDS 0
#endif

DEV void copy16_to_lds(__bf16* lds, const __bf16* g) {
#if USE_ASYNC_LDS
  __builtin_amdgcn_global_load_async_to_lds_b128((gptr_v4i)g, (lptr_v4i)lds,
                                                 0, 0);
#else
  *(uint4*)lds = *(const uint4*)g;
#endif
}

DEV void wait_async_copies() {
#if USE_ASYNC_LDS
  __asm__ volatile("s_wait_asynccnt 0" ::: "memory");
#endif
}

// Build a v16bf A/B fragment from two 16-byte LDS/global chunks.
DEV v16bf make_frag(const __bf16* p0, const __bf16* p1) {
  union { uint4 u[2]; v16bf v; } t;
  t.u[0] = *(const uint4*)p0;
  t.u[1] = *(const uint4*)p1;
  return t.v;
}

DEV v8f wmma_bf16(v16bf a, v16bf b, v8f c) {
  return __builtin_amdgcn_wmma_f32_16x16x32_bf16(
      false, a, false, b, (short)0, c, false, false);
}

// ---------------------------------------------------------------------------
// 1) f32 -> bf16 conversion (vectorized x4)
// ---------------------------------------------------------------------------
__global__ __launch_bounds__(256) void cvt_f32_bf16(
    const float* __restrict__ src, __bf16* __restrict__ dst, int n) {
  int i = (blockIdx.x * blockDim.x + threadIdx.x) * 4;
  if (i + 3 < n) {
    float4 f = *(const float4*)(src + i);
    union { __bf16 h[4]; uint2 u; } t;
    t.h[0] = (__bf16)f.x; t.h[1] = (__bf16)f.y;
    t.h[2] = (__bf16)f.z; t.h[3] = (__bf16)f.w;
    *(uint2*)(dst + i) = t.u;
  }
}

// ---------------------------------------------------------------------------
// 2/4) GEMM: C[M x 256] = A_bf16[M x 256] @ W_f32[256 x 256] + bias
// Block: 128 threads (4 waves). Block tile 64(M) x 64(N); wave = 16 x 64.
// ---------------------------------------------------------------------------
template <typename OutT>
__global__ __launch_bounds__(128) void gemm256_bf16(
    const __bf16* __restrict__ A, const float* __restrict__ W,
    const float* __restrict__ bias, OutT* __restrict__ C) {
  __shared__ __align__(16) __bf16 Al[64 * 32];   // [m][k]
  __shared__ __align__(16) __bf16 Wl[64 * 32];   // [n][k] (transposed)

  const int tid  = threadIdx.x;
  const int wave = tid >> 5, lane = tid & 31;
  const int hp   = lane >> 4, ln = lane & 15;
  const int mBase = blockIdx.x * 64;
  const int nBase = blockIdx.y * 64;

  const v8f z8 = {0.f, 0.f, 0.f, 0.f, 0.f, 0.f, 0.f, 0.f};
  v8f acc[4] = {z8, z8, z8, z8};

  for (int kb = 0; kb < HID; kb += 32) {
    __syncthreads();
    // Stage A tile: 64 rows x 32 bf16 via async 16B copies.
    for (int c = tid; c < 256; c += 128) {
      int r = c >> 2, cc = (c & 3) * 8;
      copy16_to_lds(&Al[r * 32 + cc],
                    &A[(size_t)(mBase + r) * HID + kb + cc]);
    }
    // Stage W tile transposed ([n][k]) with f32->bf16 (elementwise, VALU).
    for (int c = tid; c < 512; c += 128) {
      int kr = c >> 4, nc = (c & 15) * 4;
      float4 w = *(const float4*)&W[(size_t)(kb + kr) * HID + nBase + nc];
      Wl[(nc + 0) * 32 + kr] = (__bf16)w.x;
      Wl[(nc + 1) * 32 + kr] = (__bf16)w.y;
      Wl[(nc + 2) * 32 + kr] = (__bf16)w.z;
      Wl[(nc + 3) * 32 + kr] = (__bf16)w.w;
    }
    wait_async_copies();
    __syncthreads();

    // A fragment: row = ln, K chunks at hp*8 and 16+hp*8 (ISA 16-bit A layout)
    const __bf16* ap = &Al[(wave * 16 + ln) * 32 + hp * 8];
    v16bf af = make_frag(ap, ap + 16);
#pragma unroll
    for (int j = 0; j < 4; ++j) {
      // B fragment: col = ln, K chunks at hp*8 and 16+hp*8 in [n][k] LDS
      const __bf16* bp = &Wl[(j * 16 + ln) * 32 + hp * 8];
      acc[j] = wmma_bf16(af, make_frag(bp, bp + 16), acc[j]);
    }
  }

  // Epilogue: C/D layout -> lane row = hp*8+i, col = ln
#pragma unroll
  for (int j = 0; j < 4; ++j) {
    int col = nBase + j * 16 + ln;
    float bv = bias[col];
#pragma unroll
    for (int i = 0; i < 8; ++i) {
      int row = mBase + wave * 16 + hp * 8 + i;
      C[(size_t)row * HID + col] = (OutT)(acc[j][i] + bv);
    }
  }
}

// ---------------------------------------------------------------------------
// 3) Flash attention. Grid: (SEQ/64, NH, BATCH). Block: 128 threads (4 waves).
// Each wave owns 16 q-rows; online softmax over KV in 64-wide tiles.
// ---------------------------------------------------------------------------
__global__ __launch_bounds__(128) void flash_attn(
    const __bf16* __restrict__ Qh, const __bf16* __restrict__ Kh,
    const __bf16* __restrict__ Vh, const int* __restrict__ kv_mask,
    __bf16* __restrict__ Ob) {
  __shared__ __align__(16) __bf16 Kl[64 * 64];      // [kv][dh] == B^T for QK^T
  __shared__ __align__(16) __bf16 Vt[64 * 64];      // [dh][kv] == B^T for PV
  __shared__ __align__(16) __bf16 Pl[4 * 16 * 64];  // per-wave P stripe
  __shared__ float mb[64];                          // mask bias per kv col

  const int tid  = threadIdx.x;
  const int wave = tid >> 5, lane = tid & 31;
  const int hp   = lane >> 4, ln = lane & 15;
  const int b = blockIdx.z, h = blockIdx.y;

  // Preload Q A-fragments (row = ln within wave stripe), reused for all KV.
  const int qRow = blockIdx.x * 64 + wave * 16 + ln;
  const __bf16* qp = Qh + ((size_t)(b * SEQ + qRow)) * HID + h * DH;
  v16bf aQ[2];
#pragma unroll
  for (int ks = 0; ks < 2; ++ks)
    aQ[ks] = make_frag(qp + ks * 32 + hp * 8, qp + ks * 32 + 16 + hp * 8);

  const v8f z8 = {0.f, 0.f, 0.f, 0.f, 0.f, 0.f, 0.f, 0.f};
  v8f o[4] = {z8, z8, z8, z8};
  float mrow[8], lrow[8];
#pragma unroll
  for (int i = 0; i < 8; ++i) { mrow[i] = -3.0e38f; lrow[i] = 0.f; }

  const size_t kvBase = ((size_t)b * SEQ) * HID + (size_t)h * DH;
  constexpr float INV_SCALE = 0.125f;  // 1/sqrt(64)

  for (int kt = 0; kt < SEQ; kt += 64) {
    __syncthreads();
    // Stage K tile straight ([kv][dh]) via async 16B copies.
    for (int c = tid; c < 512; c += 128) {
      int r = c >> 3, cc = (c & 7) * 8;
      copy16_to_lds(&Kl[r * 64 + cc],
                    &Kh[kvBase + (size_t)(kt + r) * HID + cc]);
    }
    // Stage V tile transposed: Vt[dh][kv].
    for (int c = tid; c < 4096; c += 128) {
      int r = c >> 6, d = c & 63;
      Vt[d * 64 + r] = Vh[kvBase + (size_t)(kt + r) * HID + d];
    }
    if (tid < 64)
      mb[tid] = kv_mask[b * SEQ + kt + tid] ? 0.f : -1e30f;
    // Prefetch next KV tile into the cache hierarchy.
    if (kt + 64 < SEQ && tid < 64) {
      __builtin_prefetch(&Kh[kvBase + (size_t)(kt + 64 + tid) * HID], 0, 1);
      __builtin_prefetch(&Vh[kvBase + (size_t)(kt + 64 + tid) * HID], 0, 1);
    }
    wait_async_copies();
    __syncthreads();

    // S(16x64) = Q_stripe @ K^T, 2 K-steps over dh.
    v8f s[4] = {z8, z8, z8, z8};
#pragma unroll
    for (int ks = 0; ks < 2; ++ks)
#pragma unroll
      for (int j = 0; j < 4; ++j) {
        const __bf16* bp = &Kl[(j * 16 + ln) * 64 + ks * 32 + hp * 8];
        s[j] = wmma_bf16(aQ[ks], make_frag(bp, bp + 16), s[j]);
      }

    // Online softmax. Lane holds rows hp*8+i at col j*16+ln; a full row spans
    // the 16 lanes of this half-wave -> xor-shuffle reductions (1,2,4,8).
    float mbv[4];
#pragma unroll
    for (int j = 0; j < 4; ++j) mbv[j] = mb[j * 16 + ln];

#pragma unroll
    for (int i = 0; i < 8; ++i) {
      float rmax = -3.0e38f;
#pragma unroll
      for (int j = 0; j < 4; ++j) {
        float sv = s[j][i] * INV_SCALE + mbv[j];
        s[j][i] = sv;
        rmax = fmaxf(rmax, sv);
      }
#pragma unroll
      for (int d = 1; d < 16; d <<= 1)
        rmax = fmaxf(rmax, __shfl_xor(rmax, d, 32));
      float mnew = fmaxf(mrow[i], rmax);
      float f = __expf(mrow[i] - mnew);
      mrow[i] = mnew;
      float rsum = 0.f;
#pragma unroll
      for (int j = 0; j < 4; ++j) {
        float p = __expf(s[j][i] - mnew);
        rsum += p;
        // Write P (bf16) straight to the per-wave LDS stripe (A-layout src);
        // avoids keeping a 32-reg live exp array.
        Pl[wave * 1024 + (hp * 8 + i) * 64 + j * 16 + ln] = (__bf16)p;
      }
#pragma unroll
      for (int d = 1; d < 16; d <<= 1)
        rsum += __shfl_xor(rsum, d, 32);
      lrow[i] = lrow[i] * f + rsum;
#pragma unroll
      for (int jo = 0; jo < 4; ++jo) o[jo][i] *= f;
    }
    __asm__ volatile("s_wait_dscnt 0" ::: "memory");  // wave-local RAW on LDS

    // O(16x64) += P(16x64) @ V(64x64), K-dim = kv.
    const __bf16* pb = &Pl[wave * 1024 + ln * 64];
#pragma unroll
    for (int ks = 0; ks < 2; ++ks) {
      v16bf aP = make_frag(pb + ks * 32 + hp * 8, pb + ks * 32 + 16 + hp * 8);
#pragma unroll
      for (int jo = 0; jo < 4; ++jo) {
        const __bf16* vp = &Vt[(jo * 16 + ln) * 64 + ks * 32 + hp * 8];
        o[jo] = wmma_bf16(aP, make_frag(vp, vp + 16), o[jo]);
      }
    }
  }

  // Normalize and store bf16 O (head-h columns of [B*S, 256]).
#pragma unroll
  for (int jo = 0; jo < 4; ++jo)
#pragma unroll
    for (int i = 0; i < 8; ++i) {
      float l = lrow[i];
      float v = (l > 0.f) ? o[jo][i] / l : 0.f;
      int row = blockIdx.x * 64 + wave * 16 + hp * 8 + i;
      Ob[((size_t)(b * SEQ + row)) * HID + h * DH + jo * 16 + ln] = (__bf16)v;
    }
}

// ---------------------------------------------------------------------------
// Launch
// ---------------------------------------------------------------------------
extern "C" void kernel_launch(void* const* d_in, const int* in_sizes, int n_in,
                              void* d_out, int out_size, void* d_ws,
                              size_t ws_size, hipStream_t stream) {
  const float* query     = (const float*)d_in[0];
  const float* key_value = (const float*)d_in[1];
  const int*   kv_mask   = (const int*)d_in[2];
  const float* Wq = (const float*)d_in[3];
  const float* bq = (const float*)d_in[4];
  const float* Wk = (const float*)d_in[5];
  const float* bk = (const float*)d_in[6];
  const float* Wv = (const float*)d_in[7];
  const float* bv = (const float*)d_in[8];
  const float* Wo = (const float*)d_in[9];
  const float* bo = (const float*)d_in[10];

  const size_t SZ = (size_t)MROWS * HID;  // elements per [8192 x 256] matrix
  __bf16* Xq  = (__bf16*)d_ws;
  __bf16* Xkv = Xq  + SZ;
  __bf16* Qh  = Xkv + SZ;
  __bf16* Kh  = Qh  + SZ;
  __bf16* Vh  = Kh  + SZ;
  __bf16* Ob  = Xq;  // Xq dead after Q projection

  const int cvtBlocks = (int)(SZ / (256 * 4));  // 2048
  cvt_f32_bf16<<<cvtBlocks, 256, 0, stream>>>(query, Xq, (int)SZ);
  cvt_f32_bf16<<<cvtBlocks, 256, 0, stream>>>(key_value, Xkv, (int)SZ);

  dim3 gGemm(MROWS / 64, HID / 64);  // (128, 4)
  gemm256_bf16<__bf16><<<gGemm, 128, 0, stream>>>(Xq,  Wq, bq, Qh);
  gemm256_bf16<__bf16><<<gGemm, 128, 0, stream>>>(Xkv, Wk, bk, Kh);
  gemm256_bf16<__bf16><<<gGemm, 128, 0, stream>>>(Xkv, Wv, bv, Vh);

  dim3 gAttn(SEQ / 64, NH, BATCH);   // (64, 4, 2)
  flash_attn<<<gAttn, 128, 0, stream>>>(Qh, Kh, Vh, kv_mask, Ob);

  gemm256_bf16<float><<<gGemm, 128, 0, stream>>>(Ob, Wo, bo, (float*)d_out);
}
